// SynthesisBlockSparse_43104291782962
// MI455X (gfx1250) — matrix operand order repaired and checked by reading, not compiled, not run on hardware
//
#include <hip/hip_runtime.h>

typedef __attribute__((ext_vector_type(16))) _Float16     v16h;
typedef __attribute__((ext_vector_type(8)))  float        v8f;
typedef __attribute__((ext_vector_type(4)))  unsigned int v4u;

namespace {
constexpr int BB   = 4;
constexpr int R0c  = 16;
constexpr int R1c  = 32;
constexpr int CINc = 512;
constexpr int COc  = 256;
constexpr int WDIMc= 512;
constexpr int KT   = 27;
constexpr int V0c  = R0c*R0c*R0c;   // 4096
constexpr int V1c  = R1c*R1c*R1c;   // 32768
constexpr int N0c  = BB*V0c;        // 16384
constexpr int N1c  = BB*V1c;        // 131072
constexpr float SQRT2_F = 1.4142135623730951f;
constexpr float CLAMP_F = 256.0f;
constexpr float RS512   = 0.04419417382415922f; // 1/sqrt(512)
constexpr float RS256   = 0.0625f;              // 1/sqrt(256)
}

union FragH { v16h h; v4u u[2]; };

// ---------------- zero pad row for out-of-bounds taps ----------------------
__global__ void zeropad_kernel(unsigned int* __restrict__ z) {
  int i = blockIdx.x * blockDim.x + threadIdx.x;   // 1024 uints = 4KB
  z[i] = 0u;
}

// ---------------- styles: s = ws_row @ A.T / sqrt(WDIM) + b ----------------
__global__ void styles_kernel(const float* __restrict__ wsin,
                              const float* __restrict__ a0w, const float* __restrict__ a0b,
                              const float* __restrict__ a1w, const float* __restrict__ a1b,
                              const float* __restrict__ a2w, const float* __restrict__ a2b,
                              float* __restrict__ s0, float* __restrict__ s1,
                              float* __restrict__ s2) {
  int tid = blockIdx.x * blockDim.x + threadIdx.x;
  if (tid < BB*CINc) {                       // s0: (B, CIN)
    int b = tid / CINc, i = tid % CINc;
    const float* w = wsin + (size_t)(b*3 + 0)*WDIMc;
    const float* a = a0w + (size_t)i*WDIMc;
    float acc = 0.f;
    for (int t = 0; t < WDIMc; ++t) acc += w[t]*a[t];
    s0[tid] = acc*RS512 + a0b[i];
  } else if (tid < BB*CINc + BB*COc) {       // s1: (B, COUT)
    int t2 = tid - BB*CINc;
    int b = t2 / COc, o = t2 % COc;
    const float* w = wsin + (size_t)(b*3 + 1)*WDIMc;
    const float* a = a1w + (size_t)o*WDIMc;
    float acc = 0.f;
    for (int t = 0; t < WDIMc; ++t) acc += w[t]*a[t];
    s1[t2] = acc*RS512 + a1b[o];
  } else if (tid < BB*CINc + 2*BB*COc) {     // s2: (B, COUT) * 1/sqrt(COUT)
    int t2 = tid - BB*CINc - BB*COc;
    int b = t2 / COc, o = t2 % COc;
    const float* w = wsin + (size_t)(b*3 + 2)*WDIMc;
    const float* a = a2w + (size_t)o*WDIMc;
    float acc = 0.f;
    for (int t = 0; t < WDIMc; ++t) acc += w[t]*a[t];
    s2[t2] = (acc*RS512 + a2b[o]) * RS256;
  }
}

// ---------------- demodulation factors: rsqrt(sum W^2 s^2 + eps) -----------
__global__ void demod_kernel(const float* __restrict__ W0, const float* __restrict__ W1,
                             const float* __restrict__ s0, const float* __restrict__ s1,
                             float* __restrict__ d0, float* __restrict__ d1) {
  int tid = blockIdx.x * blockDim.x + threadIdx.x;
  if (tid < BB*COc) {
    int b = tid / COc, o = tid % COc;
    float acc = 0.f;
    for (int k = 0; k < KT; ++k)
      for (int i = 0; i < CINc; ++i) {
        float w = W0[((size_t)k*CINc + i)*COc + o];
        float s = s0[b*CINc + i];
        acc += (w*w)*(s*s);
      }
    d0[tid] = rsqrtf(acc + 1e-8f);
  } else if (tid < 2*BB*COc) {
    int t2 = tid - BB*COc;
    int b = t2 / COc, o = t2 % COc;
    float acc = 0.f;
    for (int k = 0; k < KT; ++k)
      for (int i = 0; i < COc; ++i) {
        float w = W1[((size_t)k*COc + i)*COc + o];
        float s = s1[b*COc + i];
        acc += (w*w)*(s*s);
      }
    d1[t2] = rsqrtf(acc + 1e-8f);
  }
}

// ---------------- modulate x_feat by s0[batch], cast to f16 ----------------
__global__ void modulate0_kernel(const float* __restrict__ x, const float* __restrict__ s0,
                                 _Float16* __restrict__ xm) {
  size_t idx = (size_t)blockIdx.x * blockDim.x + threadIdx.x;   // N0*CIN
  int n = (int)(idx >> 9), c = (int)(idx & 511);
  int b = n >> 12;                                              // /V0
  xm[idx] = (_Float16)(x[idx] * s0[b*CINc + c]);
}

// ------------- weight transpose+convert: [K][ci][co] -> f16 [K][co][ci] ----
__global__ void wtrans_kernel(const float* __restrict__ Win, _Float16* __restrict__ Wout,
                              int CI, int CO) {
  size_t idx = (size_t)blockIdx.x * blockDim.x + threadIdx.x;
  size_t total = (size_t)KT * CI * CO;
  if (idx >= total) return;
  int kcc = CI * CO;
  int k  = (int)(idx / kcc);
  int r  = (int)(idx % kcc);
  int ci = r / CO, co = r % CO;
  Wout[((size_t)k*CO + co)*CI + ci] = (_Float16)Win[idx];
}

// ---------------- WMMA gather-conv core -----------------------------------
// Wave computes a 64-node (4 M-tiles of 16 z-contiguous nodes) x 64-cout tile
// (4 N-tiles). Per K-step: 8 A-loads + 8 B-loads (all issued up front, no
// divergent branches: OOB rows read a zeroed pad row) -> 16 WMMAs.
// MODE 0: h0 = acc*d + bias (fp32, R0 grid)
// MODE 1: out = act(acc*d + bias + noise*ns)   (fp32, R1 grid)
template<int R, int CI, int MODE>
__global__ __launch_bounds__(256) void conv_kernel(
    const _Float16* __restrict__ xm, const _Float16* __restrict__ Wt,
    const _Float16* __restrict__ zrow,
    const float* __restrict__ demod, const float* __restrict__ bias,
    const float* __restrict__ noise, const float* __restrict__ nsp,
    float* __restrict__ out) {
  constexpr int V = R*R*R;
  const int lane  = threadIdx.x & 31;
  const int wid   = blockIdx.x * (blockDim.x >> 5) + (threadIdx.x >> 5);
  const int nq    = wid & 3;           // which 64-wide cout quarter
  const int ng    = wid >> 2;          // 64-node group
  const int node0 = ng * 64;
  const int nbase = nq * 64;
  const int grp   = lane >> 4;
  const int ln    = lane & 15;

  int tb[4], tx[4], ty[4], tz0[4];
#pragma unroll
  for (int t = 0; t < 4; ++t) {
    tb[t] = node0 + 16*t;
    int b = tb[t] / V;
    int rem = tb[t] - b*V;
    tx[t]  = rem / (R*R);
    ty[t]  = (rem / R) % R;
    tz0[t] = rem % R;
  }

  v8f acc[4][4];
  v8f zero = {};
#pragma unroll
  for (int t = 0; t < 4; ++t)
#pragma unroll
    for (int nt = 0; nt < 4; ++nt) acc[t][nt] = zero;

  // per-lane B row pointers (n = nbase + nt*16 + ln), advanced by k below
  const _Float16* brow[4];
#pragma unroll
  for (int nt = 0; nt < 4; ++nt)
    brow[nt] = Wt + (size_t)(nbase + nt*16 + ln)*CI;

  for (int k = 0; k < KT; ++k) {
    const int dx = k/9 - 1, dy = (k/3)%3 - 1, dz = k%3 - 1;
    const int delta = (dx*R + dy)*R + dz;
    const _Float16* arow[4];
#pragma unroll
    for (int t = 0; t < 4; ++t) {
      int xx = tx[t] + dx, yy = ty[t] + dy, zz = tz0[t] + ln + dz;
      bool ok = ((unsigned)xx < (unsigned)R) && ((unsigned)yy < (unsigned)R) &&
                ((unsigned)zz < (unsigned)R);
      // branchless: OOB lanes read the zero pad row
      arow[t] = ok ? (xm + (long)(tb[t] + ln + delta) * CI) : zrow;
    }
    const size_t kw = (size_t)k * COc * CI;   // weight slab for this tap
    for (int c0 = 0; c0 < CI; c0 += 32) {
      FragH a[4], bf[4];
#pragma unroll
      for (int t = 0; t < 4; ++t) {           // A: K = 8g..8g+7 | 16+8g..
        const _Float16* p = arow[t] + c0 + 8*grp;
        a[t].u[0] = *reinterpret_cast<const v4u*>(p);
        a[t].u[1] = *reinterpret_cast<const v4u*>(p + 16);
      }
#pragma unroll
      for (int nt = 0; nt < 4; ++nt) {        // B: n = ln, K = 8g..8g+7 | 16+8g..
        const _Float16* q = brow[nt] + kw + c0 + 8*grp;
        bf[nt].u[0] = *reinterpret_cast<const v4u*>(q);
        bf[nt].u[1] = *reinterpret_cast<const v4u*>(q + 16);
      }
#pragma unroll
      for (int t = 0; t < 4; ++t)
#pragma unroll
        for (int nt = 0; nt < 4; ++nt)
          acc[t][nt] = __builtin_amdgcn_wmma_f32_16x16x32_f16(
              false, a[t].h, false, bf[nt].h, (short)0, acc[t][nt], false, false);
    }
  }

  const int bidx = node0 / V;     // whole 64-node group is one batch
  const float ns = (MODE == 1) ? nsp[0] : 0.f;
#pragma unroll
  for (int t = 0; t < 4; ++t) {
#pragma unroll
    for (int nt = 0; nt < 4; ++nt) {
      int o = nbase + nt*16 + ln;
      float dd = demod[bidx*COc + o];
      float bo = bias[o];
#pragma unroll
      for (int j = 0; j < 8; ++j) {
        int node = node0 + t*16 + grp*8 + j;   // D: m = 8*grp + j, n = ln
        float v = acc[t][nt][j] * dd + bo;
        if (MODE == 1) {
          v += noise[(size_t)node*COc + o] * ns;
          float l = v > 0.f ? v : 0.2f*v;
          v = fminf(fmaxf(l*SQRT2_F, -CLAMP_F), CLAMP_F);
        }
        out[(size_t)node*COc + o] = v;
      }
    }
  }
}

// ------- upsample + noise + act, then modulate by s1[batch], cast f16 ------
__global__ void upact_modulate_kernel(const float* __restrict__ h0,
                                      const float* __restrict__ noise0,
                                      const float* __restrict__ ns0p,
                                      const float* __restrict__ s1,
                                      _Float16* __restrict__ xm1) {
  size_t idx = (size_t)blockIdx.x * blockDim.x + threadIdx.x;  // N1*COUT
  int m = (int)(idx >> 8), c = (int)(idx & 255);
  int b = m >> 15;
  int r = m & 32767;
  int p = (b << 12) | (((r >> 10) >> 1) << 8) | ((((r >> 5) & 31) >> 1) << 4) |
          ((r & 31) >> 1);
  float t = h0[(size_t)p*COc + c] + noise0[idx] * ns0p[0];
  float l = t > 0.f ? t : 0.2f*t;
  float a = fminf(fmaxf(l*SQRT2_F, -CLAMP_F), CLAMP_F);
  xm1[idx] = (_Float16)(a * s1[b*COc + c]);
}

// ---------------- rgb head: y = (h*s2) @ rgb_W + b, clip, + img ------------
__global__ __launch_bounds__(256) void rgb_kernel(
    const float* __restrict__ h, const float* __restrict__ s2,
    const float* __restrict__ rgbW, const float* __restrict__ rgbb,
    const float* __restrict__ img_feat, float* __restrict__ img) {
  int lane = threadIdx.x & 31;
  int wid  = blockIdx.x * (blockDim.x >> 5) + (threadIdx.x >> 5);
  for (int it = 0; it < 4; ++it) {
    int m = wid*4 + it;
    int b = m >> 15;
    float y0 = 0.f, y1 = 0.f, y2 = 0.f;
    for (int o = lane; o < COc; o += 32) {
      float hv = h[(size_t)m*COc + o] * s2[b*COc + o];
      y0 += hv * rgbW[o*3 + 0];
      y1 += hv * rgbW[o*3 + 1];
      y2 += hv * rgbW[o*3 + 2];
    }
    for (int off = 16; off > 0; off >>= 1) {
      y0 += __shfl_xor(y0, off, 32);
      y1 += __shfl_xor(y1, off, 32);
      y2 += __shfl_xor(y2, off, 32);
    }
    if (lane == 0) {
      int r = m & 32767;
      int p = (b << 12) | (((r >> 10) >> 1) << 8) |
              ((((r >> 5) & 31) >> 1) << 4) | ((r & 31) >> 1);
      y0 = fminf(fmaxf(y0 + rgbb[0], -CLAMP_F), CLAMP_F);
      y1 = fminf(fmaxf(y1 + rgbb[1], -CLAMP_F), CLAMP_F);
      y2 = fminf(fmaxf(y2 + rgbb[2], -CLAMP_F), CLAMP_F);
      img[(size_t)m*3 + 0] = img_feat[(size_t)p*3 + 0] + y0;
      img[(size_t)m*3 + 1] = img_feat[(size_t)p*3 + 1] + y1;
      img[(size_t)m*3 + 2] = img_feat[(size_t)p*3 + 2] + y2;
    }
  }
}

extern "C" void kernel_launch(void* const* d_in, const int* in_sizes, int n_in,
                              void* d_out, int out_size, void* d_ws, size_t ws_size,
                              hipStream_t stream) {
  (void)in_sizes; (void)n_in; (void)out_size; (void)ws_size;
  const float* x_feat   = (const float*)d_in[0];
  const float* img_feat = (const float*)d_in[1];
  const float* ws_in    = (const float*)d_in[2];
  const float* a0w = (const float*)d_in[3];
  const float* a0b = (const float*)d_in[4];
  const float* a1w = (const float*)d_in[5];
  const float* a1b = (const float*)d_in[6];
  const float* a2w = (const float*)d_in[7];
  const float* a2b = (const float*)d_in[8];
  const float* W0  = (const float*)d_in[9];
  const float* b0  = (const float*)d_in[10];
  const float* W1  = (const float*)d_in[11];
  const float* b1  = (const float*)d_in[12];
  const float* rgbW = (const float*)d_in[13];
  const float* rgbb = (const float*)d_in[14];
  const float* ns0  = (const float*)d_in[15];
  const float* ns1  = (const float*)d_in[16];
  const float* noise0 = (const float*)d_in[17];
  const float* noise1 = (const float*)d_in[18];
  // gather/scatter/up_idx/batch arrays (d_in[19..25]) replaced by analytic decode.

  // ---- scratch carve (all offsets 16B-aligned) ----
  char* wsb = (char*)d_ws;
  float* s0 = (float*)wsb;                 // 2048
  float* s1 = s0 + BB*CINc;                // 1024
  float* s2 = s1 + BB*COc;                 // 1024
  float* d0 = s2 + BB*COc;                 // 1024
  float* d1 = d0 + BB*COc;                 // 1024  (ends at 24576 bytes)
  _Float16* zrow = (_Float16*)(wsb + 24576);                 // 4KB zero pad row
  _Float16* xm0 = (_Float16*)(wsb + 32768);                  // N0*CIN f16   = 16 MB
  _Float16* W0t = xm0 + (size_t)N0c*CINc;                    // 27*256*512   =  7 MB
  _Float16* W1t = W0t + (size_t)KT*COc*CINc;                 // 27*256*256   = 3.5 MB
  float*    h0  = (float*)(W1t + (size_t)KT*COc*COc);        // N0*COUT f32  = 16 MB
  _Float16* xm1 = (_Float16*)(h0 + (size_t)N0c*COc);         // N1*COUT f16  = 64 MB

  float* hout   = (float*)d_out;           // N1*COUT
  float* imgout = hout + (size_t)N1c*COc;  // N1*3

  // 0) zero the OOB pad row
  zeropad_kernel<<<4, 256, 0, stream>>>((unsigned int*)zrow);

  // 1) styles (tiny GEMMs) and demod factors
  styles_kernel<<<16, 256, 0, stream>>>(ws_in, a0w, a0b, a1w, a1b, a2w, a2b,
                                        s0, s1, s2);
  demod_kernel<<<8, 256, 0, stream>>>(W0, W1, s0, s1, d0, d1);

  // 2) f16 conversions: modulated activations + transposed weights
  modulate0_kernel<<<(N0c*CINc)/256, 256, 0, stream>>>(x_feat, s0, xm0);
  wtrans_kernel<<<(KT*CINc*COc + 255)/256, 256, 0, stream>>>(W0, W0t, CINc, COc);
  wtrans_kernel<<<(KT*COc*COc + 255)/256, 256, 0, stream>>>(W1, W1t, COc, COc);

  // 3) conv0: (16384/64) node-groups x 4 cout-quarters = 1024 waves = 128 blocks
  conv_kernel<R0c, CINc, 0><<<128, 256, 0, stream>>>(xm0, W0t, zrow, d0, b0,
                                                     nullptr, nullptr, h0);

  // 4) upsample + noise + act + s1 modulation -> f16 conv1 input
  upact_modulate_kernel<<<(N1c*COc)/256, 256, 0, stream>>>(h0, noise0, ns0,
                                                           s1, xm1);

  // 5) conv1: (131072/64) x 4 = 8192 waves = 1024 blocks; fused noise+act
  conv_kernel<R1c, COc, 1><<<1024, 256, 0, stream>>>(xm1, W1t, zrow, d1, b1,
                                                     noise1, ns1, hout);

  // 6) rgb head + img residual
  rgb_kernel<<<N1c/32, 256, 0, stream>>>(hout, s2, rgbW, rgbb, img_feat, imgout);
}
